// RecurrentAE_57140244906398
// MI455X (gfx1250) — matrix operand order repaired and checked by reading, not compiled, not run on hardware
//
#include <hip/hip_runtime.h>
#include <hip/hip_bf16.h>

// RecurrentAE: B=512, T=512, F=32, H=128
// Persistent per-batch-tile kernel: encoder LSTM (512 steps) then decoder
// LSTMCell+Linear (512 steps), all state on-chip, bf16 WMMA w/ f32 accum.
// Epilogue uses CDNA5 v_tanh_f32 (sigmoid via 0.5*tanh(0.5x)+0.5), biases
// folded into the epilogue FMAs so WMMA accumulators start from inline 0.

#define B_  512
#define T_  512
#define F_  32
#define H_  128
#define BT  16          // batch rows per workgroup (= WMMA M)
#define RS  176         // padded LDS row stride (F_+H_ = 160 -> 176, keeps 16B align)

typedef __attribute__((ext_vector_type(16))) __bf16 bf16x16;
typedef __attribute__((ext_vector_type(8)))  __bf16 bf16x8;
typedef __attribute__((ext_vector_type(8)))  float  f32x8;

// CDNA5 hardware tanh (TRANS32 op, co-executes with WMMA); guarded fallbacks
// avoid precise-IEEE division chains except the final __expf path.
__device__ __forceinline__ float tanh_hw(float x) {
#if __has_builtin(__builtin_amdgcn_tanhf)
    return __builtin_amdgcn_tanhf(x);
#elif __has_builtin(__builtin_amdgcn_exp2f) && __has_builtin(__builtin_amdgcn_rcpf)
    const float e = __builtin_amdgcn_exp2f(2.885390082f * x);   // exp(2x)
    return 1.0f - 2.0f * __builtin_amdgcn_rcpf(e + 1.0f);
#else
    return 1.0f - 2.0f / (__expf(2.0f * x) + 1.0f);
#endif
}

__device__ __forceinline__ f32x8 wmma_bf16(bf16x16 a, bf16x16 b, f32x8 c) {
    // D = A(16x32 bf16) * B(32x16 bf16) + C(16x16 f32)
    return __builtin_amdgcn_wmma_f32_16x16x32_bf16(false, a, false, b, (short)0, c, false, false);
}

// A-fragment (16x32 bf16) from row-major LDS staging tile.
// lane m<16 -> row m, K {kb+0..15}; lane m+16 -> K {kb+16..31}.
__device__ __forceinline__ bf16x16 load_A(const __bf16* hx, int m, int kbase, int half) {
    const __bf16* row = hx + m * RS;
    const bf16x8 lo = *(const bf16x8*)(row + kbase + half * 16);
    const bf16x8 hi = *(const bf16x8*)(row + kbase + half * 16 + 8);
    bf16x16 a;
    #pragma unroll
    for (int e = 0; e < 8; ++e) { a[e] = lo[e]; a[e + 8] = hi[e]; }
    return a;
}

// Register-resident B-fragments: Bf[kt][gate]; K axis = concat [x(32) | h(128)],
// so rows of Wih/Whh are contiguous in global memory.
__device__ __forceinline__ void load_lstm_B(bf16x16 Bf[5][4], float bias[4],
                                            const float* __restrict__ Wih,
                                            const float* __restrict__ Whh,
                                            const float* __restrict__ bih,
                                            const float* __restrict__ bhh,
                                            int w, int nc, int half) {
    #pragma unroll
    for (int gate = 0; gate < 4; ++gate) {
        const int g = gate * H_ + 16 * w + nc;     // gate column owned by this lane
        bias[gate] = bih[g] + bhh[g];
        #pragma unroll
        for (int kt = 0; kt < 5; ++kt) {
            #pragma unroll
            for (int e = 0; e < 16; ++e) {
                const int k = kt * 32 + half * 16 + e;
                const float v = (k < F_) ? Wih[g * F_ + k] : Whh[g * H_ + (k - F_)];
                Bf[kt][gate][e] = (__bf16)v;
            }
        }
    }
}

#define ZERO8 {0.0f,0.0f,0.0f,0.0f,0.0f,0.0f,0.0f,0.0f}

__global__ __launch_bounds__(256, 1)
void recurrent_ae_57140244906398_kernel(const float* __restrict__ x,
                                        const float* __restrict__ eWih, const float* __restrict__ eWhh,
                                        const float* __restrict__ ebih, const float* __restrict__ ebhh,
                                        const float* __restrict__ dWih, const float* __restrict__ dWhh,
                                        const float* __restrict__ dbih, const float* __restrict__ dbhh,
                                        const float* __restrict__ dnW,  const float* __restrict__ dnB,
                                        float* __restrict__ out)
{
    // Staging tile: cols [0,32) = current LSTM input x_t, cols [32,160) = h.
    __shared__ __align__(16) __bf16 hx[BT * RS];

    const int tid  = threadIdx.x;
    const int lane = tid & 31;
    const int w    = tid >> 5;       // 8 waves; wave w owns hidden cols [16w, 16w+16)
    const int nc   = lane & 15;
    const int half = lane >> 4;
    const int b0   = blockIdx.x * BT;
    const int mrow = 8 * half;       // D-fragment: element r <-> batch row r + 8*half

    bf16x16 Bf[5][4];
    float   bias[4];
    f32x8   cfrag = ZERO8;           // cell state lives in registers for all 1024 steps

    // ======================= encoder =======================
    load_lstm_B(Bf, bias, eWih, eWhh, ebih, ebhh, w, nc, half);
    // half-biases for sigmoid-via-tanh: sigm(z+b) = 0.5*tanh(0.5z+0.5b)+0.5
    float hbi = 0.5f * bias[0], hbf = 0.5f * bias[1], bg = bias[2], hbo = 0.5f * bias[3];

    for (int idx = tid; idx < BT * H_; idx += 256)
        hx[(idx >> 7) * RS + F_ + (idx & 127)] = (__bf16)0.0f;   // h0 = 0

    for (int t = 0; t < T_; ++t) {
        // stage x_t tile (16 x 32 fp32 -> bf16), coalesced 128B rows
        for (int idx = tid; idx < BT * F_; idx += 256) {
            const int m = idx >> 5, c = idx & 31;
            hx[m * RS + c] = (__bf16)x[(size_t)(b0 + m) * (T_ * F_) + (size_t)t * F_ + c];
        }
        if (t + 1 < T_)   // gfx1250 global_prefetch_b8 for next step's rows
            __builtin_prefetch(&x[(size_t)(b0 + (tid & 15)) * (T_ * F_) + (size_t)(t + 1) * F_], 0, 0);
        __syncthreads();   // x_t and h_{t-1} ready

        f32x8 zi = ZERO8, zf = ZERO8, zg = ZERO8, zo = ZERO8;  // C starts as inline 0
        #pragma unroll
        for (int kt = 0; kt < 5; ++kt) {   // K = [x|h] = 160 -> 5 tiles, 20 WMMAs
            const bf16x16 a = load_A(hx, nc, kt * 32, half);
            zi = wmma_bf16(a, Bf[kt][0], zi);
            zf = wmma_bf16(a, Bf[kt][1], zf);
            zg = wmma_bf16(a, Bf[kt][2], zg);
            zo = wmma_bf16(a, Bf[kt][3], zo);
        }
        __syncthreads();   // all waves done reading hx (WAR on h region)

        #pragma unroll
        for (int r = 0; r < 8; ++r) {
            const float si = 0.5f * tanh_hw(0.5f * zi[r] + hbi) + 0.5f;
            const float sf = 0.5f * tanh_hw(0.5f * zf[r] + hbf) + 0.5f;
            const float gg = tanh_hw(zg[r] + bg);
            const float so = 0.5f * tanh_hw(0.5f * zo[r] + hbo) + 0.5f;
            const float cn = sf * cfrag[r] + si * gg;
            cfrag[r] = cn;
            hx[(mrow + r) * RS + F_ + 16 * w + nc] = (__bf16)(so * tanh_hw(cn));
        }
    }
    __syncthreads();       // (h_n, c_n) complete; c_n stays in cfrag

    // ======================= decoder =======================
    load_lstm_B(Bf, bias, dWih, dWhh, dbih, dbhh, w, nc, half);
    hbi = 0.5f * bias[0]; hbf = 0.5f * bias[1]; bg = bias[2]; hbo = 0.5f * bias[3];

    bf16x16 Bd[4];         // dense_W.T fragments: only waves 0,1 (N = 32 -> 2 tiles)
    float dbias = 0.0f;
    if (w < 2) {
        const int col = 16 * w + nc;
        dbias = dnB[col];
        #pragma unroll
        for (int kt = 0; kt < 4; ++kt)
            #pragma unroll
            for (int e = 0; e < 16; ++e)
                Bd[kt][e] = (__bf16)dnW[col * H_ + kt * 32 + half * 16 + e];

        // x_i0 = h_n @ dense_W.T + dense_b   (feeds step 0, not an output)
        f32x8 xo = ZERO8;
        #pragma unroll
        for (int kt = 0; kt < 4; ++kt)
            xo = wmma_bf16(load_A(hx, nc, F_ + kt * 32, half), Bd[kt], xo);
        #pragma unroll
        for (int r = 0; r < 8; ++r)
            hx[(mrow + r) * RS + col] = (__bf16)(xo[r] + dbias);
    }
    __syncthreads();

    for (int t = 0; t < T_; ++t) {
        f32x8 zi = ZERO8, zf = ZERO8, zg = ZERO8, zo = ZERO8;
        #pragma unroll
        for (int kt = 0; kt < 5; ++kt) {
            const bf16x16 a = load_A(hx, nc, kt * 32, half);
            zi = wmma_bf16(a, Bf[kt][0], zi);
            zf = wmma_bf16(a, Bf[kt][1], zf);
            zg = wmma_bf16(a, Bf[kt][2], zg);
            zo = wmma_bf16(a, Bf[kt][3], zo);
        }
        __syncthreads();   // done reading hx

        #pragma unroll
        for (int r = 0; r < 8; ++r) {
            const float si = 0.5f * tanh_hw(0.5f * zi[r] + hbi) + 0.5f;
            const float sf = 0.5f * tanh_hw(0.5f * zf[r] + hbf) + 0.5f;
            const float gg = tanh_hw(zg[r] + bg);
            const float so = 0.5f * tanh_hw(0.5f * zo[r] + hbo) + 0.5f;
            const float cn = sf * cfrag[r] + si * gg;
            cfrag[r] = cn;
            hx[(mrow + r) * RS + F_ + 16 * w + nc] = (__bf16)(so * tanh_hw(cn));
        }
        __syncthreads();   // h_t visible

        if (w < 2) {       // x_{t+1} = h_t @ dense_W.T + b ; also the output (flipped)
            f32x8 xo = ZERO8;
            #pragma unroll
            for (int kt = 0; kt < 4; ++kt)
                xo = wmma_bf16(load_A(hx, nc, F_ + kt * 32, half), Bd[kt], xo);
            const int col = 16 * w + nc;
            #pragma unroll
            for (int r = 0; r < 8; ++r) {
                const float v = xo[r] + dbias;
                out[(size_t)(b0 + mrow + r) * (T_ * F_) + (size_t)(T_ - 1 - t) * F_ + col] = v;
                hx[(mrow + r) * RS + col] = (__bf16)v;
            }
        }
        __syncthreads();   // x region visible for next step
    }
}

extern "C" void kernel_launch(void* const* d_in, const int* in_sizes, int n_in,
                              void* d_out, int out_size, void* d_ws, size_t ws_size,
                              hipStream_t stream) {
    (void)in_sizes; (void)n_in; (void)d_ws; (void)ws_size; (void)out_size;
    recurrent_ae_57140244906398_kernel<<<dim3(B_ / BT), dim3(256), 0, stream>>>(
        (const float*)d_in[0],
        (const float*)d_in[1], (const float*)d_in[2],
        (const float*)d_in[3], (const float*)d_in[4],
        (const float*)d_in[5], (const float*)d_in[6],
        (const float*)d_in[7], (const float*)d_in[8],
        (const float*)d_in[9], (const float*)d_in[10],
        (float*)d_out);
}
